// LocalAttn_48180943127216
// MI455X (gfx1250) — matrix operand order repaired and checked by reading, not compiled, not run on hardware
//
#include <hip/hip_runtime.h>
#include <hip/hip_bf16.h>
#include <stdint.h>

// ---------------------------------------------------------------------------
// Problem constants (from reference): B=16, 56x56, HDIM=512, HEADS=8, PS=7
// ---------------------------------------------------------------------------
#define IMG_H     56
#define IMG_W     56
#define N_TOK     3136          // 56*56
#define HD        512
#define HD3       1536
#define N_HEADS   8
#define DH        64            // 512/8
#define WIN       7
#define KTOK      49            // 7*7 tokens per window
#define NWIN      64            // 8*8 windows
#define BATCH     16
#define M_ROWS    (BATCH * N_TOK)   // 50176

typedef __attribute__((ext_vector_type(16))) __bf16 v16bf;
typedef __attribute__((ext_vector_type(8)))  float  v8f;

#define WMMA_BF16(a, b, c) \
  __builtin_amdgcn_wmma_f32_16x16x32_bf16(false, (a), false, (b), (short)0, (c), false, false)

struct alignas(8) bf16x4 { __bf16 a, b, c, d; };

// ---------------------------------------------------------------------------
// Fragment loaders (wave32, contiguous k = 16*(lane/16) + i, same for A and B)
// A 16x32 bf16: lane L<16 -> M=L, K = 0..15; lane>=16 -> M=L-16, K = 16..31
// ---------------------------------------------------------------------------
static __device__ __forceinline__ v16bf load_a_frag(const __bf16* base, int stride, int lane) {
  const int m  = lane & 15;
  const int kh = (lane >> 4) * 16;
  const __bf16* rp = base + m * stride;
  v16bf a;
#pragma unroll
  for (int i = 0; i < 8; ++i) a[i]     = rp[kh + i];
#pragma unroll
  for (int i = 0; i < 8; ++i) a[8 + i] = rp[8 + kh + i];
  return a;
}

// B 32x16 bf16 from row-major (K rows x N cols) source: lane L<16 holds K=0..15
// of column N=L; lane>=16 holds K=16..31 of column N=L-16.
static __device__ __forceinline__ v16bf load_b_frag_rm(const __bf16* base, int stride, int lane) {
  const int n  = lane & 15;
  const int k0 = (lane >> 4) * 16;
  v16bf b;
#pragma unroll
  for (int i = 0; i < 16; ++i) b[i] = base[(k0 + i) * stride + n];
  return b;
}

// B 32x16 where B(k,n) = src[n][k]  (for S = Q @ K^T : src rows are keys)
static __device__ __forceinline__ v16bf load_b_frag_tr(const __bf16* base, int stride, int lane) {
  const int n  = lane & 15;
  const int k0 = (lane >> 4) * 16;
  const __bf16* rp = base + n * stride;
  v16bf b;
#pragma unroll
  for (int i = 0; i < 16; ++i) b[i] = rp[k0 + i];
  return b;
}

static __device__ __forceinline__ v8f vzero8() {
  v8f z;
#pragma unroll
  for (int i = 0; i < 8; ++i) z[i] = 0.0f;
  return z;
}

// ---------------------------------------------------------------------------
// GEMM staging helpers: global -> registers -> (bf16-converted) LDS
// A tile: 64x32, B tile: 32x64, 128 threads, 4 chunks/thread each.
// ---------------------------------------------------------------------------
template <bool A_BF16>
static __device__ __forceinline__ void stage_load(
    const void* __restrict__ Ap, const float* __restrict__ Bp,
    int K, int N, int m0, int n0, int k0, int tid,
    float4* aR, uint2* aR16, float4* bR) {
  if (A_BF16) {
    const __bf16* A = (const __bf16*)Ap;
#pragma unroll
    for (int j = 0; j < 4; ++j) {
      const int id = tid + 128 * j, r = id >> 3, c4 = id & 7;
      aR16[j] = *(const uint2*)(A + (size_t)(m0 + r) * K + k0 + c4 * 4);
    }
  } else {
    const float* A = (const float*)Ap;
#pragma unroll
    for (int j = 0; j < 4; ++j) {
      const int id = tid + 128 * j, r = id >> 3, c4 = id & 7;
      aR[j] = *(const float4*)(A + (size_t)(m0 + r) * K + k0 + c4 * 4);
    }
  }
#pragma unroll
  for (int j = 0; j < 4; ++j) {
    const int id = tid + 128 * j, r = id >> 4, c4 = id & 15;
    bR[j] = *(const float4*)(Bp + (size_t)(k0 + r) * N + n0 + c4 * 4);
  }
}

template <bool A_BF16>
static __device__ __forceinline__ void stage_store(
    __bf16 (*As)[48], __bf16 (*Bs)[72], int tid,
    const float4* aR, const uint2* aR16, const float4* bR) {
  if (A_BF16) {
#pragma unroll
    for (int j = 0; j < 4; ++j) {
      const int id = tid + 128 * j, r = id >> 3, c4 = id & 7;
      *(uint2*)(&As[r][c4 * 4]) = aR16[j];
    }
  } else {
#pragma unroll
    for (int j = 0; j < 4; ++j) {
      const int id = tid + 128 * j, r = id >> 3, c4 = id & 7;
      const bf16x4 h{(__bf16)aR[j].x, (__bf16)aR[j].y, (__bf16)aR[j].z, (__bf16)aR[j].w};
      *(bf16x4*)(&As[r][c4 * 4]) = h;
    }
  }
#pragma unroll
  for (int j = 0; j < 4; ++j) {
    const int id = tid + 128 * j, r = id >> 4, c4 = id & 15;
    const bf16x4 h{(__bf16)bR[j].x, (__bf16)bR[j].y, (__bf16)bR[j].z, (__bf16)bR[j].w};
    *(bf16x4*)(&Bs[r][c4 * 4]) = h;
  }
}

// ---------------------------------------------------------------------------
// Tiled GEMM:  C[M,N] = A[M,512] * B[512,N] (+bias).  fp32->bf16 fused into
// staging. BM=64, BN=64, BK=32, double-buffered LDS, 128 threads = 4 waves,
// each wave computing a 16x64 strip (4 WMMA per K-step).
// ---------------------------------------------------------------------------
template <bool A_BF16, bool OUT_BF16, bool ADD_BIAS>
__global__ __launch_bounds__(128) void gemm_wmma_kernel(
    const void* __restrict__ Ap, const float* __restrict__ Bp,
    void* __restrict__ Cp, const float* __restrict__ bias,
    int N, int K) {
  __shared__ __bf16 As[2][64][48];   // 64 x 32 (+16 pad) per buffer
  __shared__ __bf16 Bs[2][32][72];   // 32 x 64 (+8 pad)  per buffer
  __shared__ __bf16 Cs[64][72];      // bf16 epilogue staging

  const int tid  = threadIdx.x;
  const int lane = tid & 31;
  const int wave = tid >> 5;
  const int n0 = blockIdx.x * 64;
  const int m0 = blockIdx.y * 64;
  const int NT = K / 32;

  v8f acc[4];
#pragma unroll
  for (int nt = 0; nt < 4; ++nt) acc[nt] = vzero8();

  float4 aR[4];
  uint2  aR16[4];
  float4 bR[4];

  // prologue: stage tile 0
  stage_load<A_BF16>(Ap, Bp, K, N, m0, n0, 0, tid, aR, aR16, bR);
  stage_store<A_BF16>(As[0], Bs[0], tid, aR, aR16, bR);
  __syncthreads();

  for (int kt = 0; kt < NT; ++kt) {
    const int cur = kt & 1;
    const bool more = (kt + 1 < NT);

    // issue next tile's global loads before consuming the current tile
    if (more)
      stage_load<A_BF16>(Ap, Bp, K, N, m0, n0, (kt + 1) * 32, tid, aR, aR16, bR);
    if (kt + 2 < NT) {   // warm L2/L0 two tiles ahead (global_prefetch_b8)
      const int kp = (kt + 2) * 32;
      __builtin_prefetch(Bp + (size_t)(kp + (tid >> 2)) * N + n0, 0, 1);
      if (A_BF16)
        __builtin_prefetch((const __bf16*)Ap + (size_t)(m0 + (tid >> 1)) * K + kp, 0, 1);
      else
        __builtin_prefetch((const float*)Ap + (size_t)(m0 + (tid >> 1)) * K + kp, 0, 1);
    }

    const v16bf afrag = load_a_frag(&As[cur][wave * 16][0], 48, lane);
#pragma unroll
    for (int nt = 0; nt < 4; ++nt) {
      const v16bf bfrag = load_b_frag_rm(&Bs[cur][0][nt * 16], 72, lane);
      acc[nt] = WMMA_BF16(afrag, bfrag, acc[nt]);
    }

    if (more)
      stage_store<A_BF16>(As[cur ^ 1], Bs[cur ^ 1], tid, aR, aR16, bR);
    __syncthreads();
  }

  // ---- epilogue ----
  const int mrl = wave * 16 + ((lane >> 4) * 8);   // local row
  const int nc  = lane & 15;
  if (OUT_BF16) {
    // stage through LDS for fully coalesced 16B stores
#pragma unroll
    for (int nt = 0; nt < 4; ++nt)
#pragma unroll
      for (int r = 0; r < 8; ++r) Cs[mrl + r][nt * 16 + nc] = (__bf16)acc[nt][r];
    __syncthreads();
    __bf16* C = (__bf16*)Cp;
#pragma unroll
    for (int j = 0; j < 4; ++j) {
      const int id = tid + 128 * j, r = id >> 3, c8 = id & 7;
      const uint4 v = *(const uint4*)(&Cs[r][c8 * 8]);
      *(uint4*)(C + (size_t)(m0 + r) * N + n0 + c8 * 8) = v;
    }
  } else {
    float* C = (float*)Cp;
#pragma unroll
    for (int nt = 0; nt < 4; ++nt) {
      const int col = n0 + nt * 16 + nc;
      const float bv = ADD_BIAS ? bias[col] : 0.0f;
#pragma unroll
      for (int r = 0; r < 8; ++r)
        C[(size_t)(m0 + mrl + r) * N + col] = acc[nt][r] + bv;
    }
  }
}

// ---------------------------------------------------------------------------
// Windowed attention: one workgroup per (window p, head g, batch b).
// Q,K,V: 49x64 zero-padded to 64x64 bf16 in LDS. 128 threads = 4 waves,
// each wave computes a 16-row strip of S (16x64) and of the output.
// ---------------------------------------------------------------------------
__global__ __launch_bounds__(128) void attn_win_kernel(
    const __bf16* __restrict__ qkv, const float* __restrict__ rel_bias,
    const unsigned char* __restrict__ mask, __bf16* __restrict__ out) {
  __shared__ __bf16 Qs[64][72];
  __shared__ __bf16 Ks[64][72];
  __shared__ __bf16 Vs[64][72];
  __shared__ float  Sf[64][72];
  __shared__ __bf16 Sb[64][72];
  __shared__ unsigned char colmask[64];

  const int p = blockIdx.x;     // window 0..63
  const int g = blockIdx.y;     // head 0..7
  const int b = blockIdx.z;     // batch 0..15
  const int wh = p >> 3, ww = p & 7;
  const int tid  = threadIdx.x;
  const int lane = tid & 31;
  const int wave = tid >> 5;
  const float scale = 0.125f;   // dh^-0.5 = 1/8

  // ---- load Q/K/V tiles (bf16, 16B vectors, zero-pad rows 49..63) ----
#pragma unroll
  for (int mat = 0; mat < 3; ++mat) {
    __bf16(*dst)[72] = (mat == 0) ? Qs : (mat == 1) ? Ks : Vs;
#pragma unroll
    for (int j = 0; j < 4; ++j) {
      const int idx = tid + 128 * j;
      const int r = idx >> 3, c8 = idx & 7;   // 8 chunks of 8 bf16 per row
      uint4 v;
      if (r < KTOK) {
        const int ph = r / WIN, pw = r % WIN;
        const int hw = (wh * WIN + ph) * IMG_W + (ww * WIN + pw);
        v = *(const uint4*)(qkv + ((size_t)b * N_TOK + hw) * HD3 + mat * HD + g * DH + c8 * 8);
      } else {
        v.x = v.y = v.z = v.w = 0u;
      }
      *(uint4*)(&dst[r][c8 * 8]) = v;
    }
  }
  if (tid < 64) {
    unsigned char m = 1;  // padded keys are masked
    if (tid < KTOK) {
      const int ph = tid / WIN, pw = tid % WIN;
      m = mask[(wh * WIN + ph) * IMG_W + (ww * WIN + pw)];
    }
    colmask[tid] = m;
  }
  __syncthreads();

  // ---- S = Q @ K^T  (each wave: rows wave*16 .. +15) ----
  {
    v8f acc[4];
#pragma unroll
    for (int nt = 0; nt < 4; ++nt) acc[nt] = vzero8();
#pragma unroll
    for (int kk = 0; kk < DH; kk += 32) {
      const v16bf a = load_a_frag(&Qs[wave * 16][kk], 72, lane);
#pragma unroll
      for (int nt = 0; nt < 4; ++nt) {
        const v16bf bb = load_b_frag_tr(&Ks[nt * 16][kk], 72, lane);
        acc[nt] = WMMA_BF16(a, bb, acc[nt]);
      }
    }
    const int mrow = wave * 16 + ((lane >> 4) * 8);
    const int nc   = lane & 15;
#pragma unroll
    for (int nt = 0; nt < 4; ++nt)
#pragma unroll
      for (int r = 0; r < 8; ++r) Sf[mrow + r][nt * 16 + nc] = acc[nt][r];
  }
  __syncthreads();

  // ---- bias + scale + mask + row softmax (threads 0..63 own one row) ----
  if (tid < 64) {
    const int q = tid;
    const float* rb = rel_bias + ((size_t)g * KTOK + (q < KTOK ? q : 0)) * KTOK;
    const float NEG_INF = -__builtin_inff();
    float mx = NEG_INF;
#pragma unroll 1
    for (int k = 0; k < 64; ++k) {
      float v = NEG_INF;
      if (q < KTOK && k < KTOK && !colmask[k]) v = (Sf[q][k] + rb[k]) * scale;
      Sf[q][k] = v;
      mx = fmaxf(mx, v);
    }
    float sum = 0.0f;
#pragma unroll 1
    for (int k = 0; k < 64; ++k) {
      const float s = Sf[q][k];
      const float e = (s == NEG_INF) ? 0.0f : __expf(s - mx);
      Sf[q][k] = e;
      sum += e;
    }
    const float inv = (sum > 0.0f) ? (1.0f / sum) : 0.0f;
#pragma unroll 1
    for (int k = 0; k < 64; ++k) Sb[q][k] = (__bf16)(Sf[q][k] * inv);
  }
  __syncthreads();

  // ---- O = P @ V ----
  v8f oacc[4];
#pragma unroll
  for (int nt = 0; nt < 4; ++nt) oacc[nt] = vzero8();
#pragma unroll
  for (int kk = 0; kk < 64; kk += 32) {
    const v16bf a = load_a_frag(&Sb[wave * 16][kk], 72, lane);
#pragma unroll
    for (int nt = 0; nt < 4; ++nt) {
      const v16bf bb = load_b_frag_rm(&Vs[kk][nt * 16], 72, lane);
      oacc[nt] = WMMA_BF16(a, bb, oacc[nt]);
    }
  }
  __syncthreads();   // all waves finished reading Sb as fragments

  // stage O tile into LDS (reuse Sb), then coalesced 16B global stores
  {
    const int mrow = wave * 16 + ((lane >> 4) * 8);
    const int nc   = lane & 15;
#pragma unroll
    for (int nt = 0; nt < 4; ++nt)
#pragma unroll
      for (int r = 0; r < 8; ++r) Sb[mrow + r][nt * 16 + nc] = (__bf16)oacc[nt][r];
  }
  __syncthreads();
  for (int idx = tid; idx < KTOK * 8; idx += 128) {
    const int q = idx >> 3, c8 = idx & 7;
    const int ph = q / WIN, pw = q % WIN;
    const int hw = (wh * WIN + ph) * IMG_W + (ww * WIN + pw);
    const uint4 v = *(const uint4*)(&Sb[q][c8 * 8]);
    *(uint4*)(out + ((size_t)b * N_TOK + hw) * HD + g * DH + c8 * 8) = v;
  }
}

// ---------------------------------------------------------------------------
// Launch: GEMM(qkv, bf16 out) -> window attention (bf16 out) -> GEMM(proj+bias)
// ---------------------------------------------------------------------------
extern "C" void kernel_launch(void* const* d_in, const int* in_sizes, int n_in,
                              void* d_out, int out_size, void* d_ws, size_t ws_size,
                              hipStream_t stream) {
  (void)in_sizes; (void)n_in; (void)out_size; (void)ws_size;
  const float*         x     = (const float*)d_in[0];
  const unsigned char* mask  = (const unsigned char*)d_in[1];
  const float*         Wqkv  = (const float*)d_in[2];
  const float*         Wproj = (const float*)d_in[3];
  const float*         bproj = (const float*)d_in[4];
  const float*         relb  = (const float*)d_in[5];
  float* out = (float*)d_out;

  char* ws = (char*)d_ws;
  __bf16* qkv_ws  = (__bf16*)ws;                                  // 50176*1536 bf16
  __bf16* attn_ws = (__bf16*)(ws + (size_t)M_ROWS * HD3 * 2);     // 50176*512 bf16

  const dim3 blk(128);

  // 1) qkv = x @ Wqkv  (fp32 in, bf16 out)
  gemm_wmma_kernel<false, true, false>
      <<<dim3(HD3 / 64, M_ROWS / 64), blk, 0, stream>>>(x, Wqkv, qkv_ws, nullptr, HD3, HD);

  // 2) windowed multi-head attention (bf16 in/out)
  attn_win_kernel<<<dim3(NWIN, N_HEADS, BATCH), blk, 0, stream>>>(qkv_ws, relb, mask, attn_ws);

  // 3) out = attn_out @ Wproj + bproj  (bf16 A, fp32 out)
  gemm_wmma_kernel<true, false, true>
      <<<dim3(HD / 64, M_ROWS / 64), blk, 0, stream>>>(attn_ws, Wproj, out, bproj, HD, HD);
}
